// MixHopVolatilityNet_84207128805734
// MI455X (gfx1250) — matrix operand order repaired and checked by reading, not compiled, not run on hardware
//
#include <hip/hip_runtime.h>

typedef unsigned int   u32;
typedef unsigned short u16;

typedef __attribute__((ext_vector_type(16))) __bf16 v16bf;
typedef __attribute__((ext_vector_type(8)))  float  v8f;

#define NN   10000
#define EE   160000
#define F_IN 84
#define KIN  96      // padded K for input gemm
#define HH   256
#define PP   4
#define PH   1024
#define MPAD 10112   // 79 * 128

__device__ __forceinline__ float gelu_f(float x) {
    return 0.5f * x * (1.0f + erff(x * 0.70710678118654752440f));
}

__device__ __forceinline__ u16 f2bf(float f) {
    u32 u = __float_as_uint(f);
    u32 r = (u + 0x7FFFu + ((u >> 16) & 1u)) >> 16;   // RNE
    return (u16)r;
}

// pack two floats to packed bf16 (round-half-up; one v_perm_b32 on gfx1250)
__device__ __forceinline__ u32 pack_bf16(float x, float y) {
#if __has_builtin(__builtin_amdgcn_cvt_pk_bf16_f32)
    typedef __attribute__((ext_vector_type(2))) __bf16 v2bf;
    union { v2bf v; u32 u; } r;
    r.v = __builtin_amdgcn_cvt_pk_bf16_f32(x, y);
    return r.u;
#elif __has_builtin(__builtin_amdgcn_perm)
    u32 ux = __float_as_uint(x) + 0x8000u;
    u32 uy = __float_as_uint(y) + 0x8000u;
    return __builtin_amdgcn_perm(uy, ux, 0x07060302u);  // {uy.b3,uy.b2,ux.b3,ux.b2}
#else
    u32 ux = __float_as_uint(x) + 0x8000u;
    u32 uy = __float_as_uint(y) + 0x8000u;
    return (uy & 0xFFFF0000u) | (ux >> 16);
#endif
}

// ---------------------------------------------------------------- small utils
__global__ void k_fill1(float* p, int n) {
    int i = blockIdx.x * 256 + threadIdx.x;
    if (i < n) p[i] = 1.0f;
}
__global__ void k_zero(float* p, int n) {
    int i = blockIdx.x * 256 + threadIdx.x;
    if (i < n) p[i] = 0.0f;
}
__global__ void k_degcnt(const int* __restrict__ dst, float* deg, int e) {
    int i = blockIdx.x * 256 + threadIdx.x;
    if (i < e) unsafeAtomicAdd(&deg[dst[i]], 1.0f);
}
__global__ void k_rsqrt(float* p, int n) {
    int i = blockIdx.x * 256 + threadIdx.x;
    if (i < n) p[i] = rsqrtf(p[i]);
}
// pack x [NN,84] -> Xp [MPAD,96] fp32, zero padded
__global__ void k_pack_x(const float* __restrict__ x, float* __restrict__ Xp) {
    int idx = blockIdx.x * 256 + threadIdx.x;
    if (idx >= MPAD * KIN) return;
    int r = idx / KIN, c = idx - r * KIN;
    Xp[idx] = (r < NN && c < F_IN) ? x[r * F_IN + c] : 0.0f;
}
// transpose+convert one weight matrix: in [K][Nout] f32 -> out [Nout][Kpad] bf16 (K..Kpad zero)
// grid: (cdiv(Kpad,256), Nout)
__global__ void k_cvt_tr(const float* __restrict__ in, u16* __restrict__ out,
                         int K, int Kpad, int Nout) {
    int k = blockIdx.x * 256 + threadIdx.x;
    int n = blockIdx.y;
    if (k >= Kpad) return;
    out[(size_t)n * Kpad + k] = (k < K) ? f2bf(in[(size_t)k * Nout + n]) : (u16)0;
}

// ---------------------------------------------------------------- SpMM (gcn propagate)
// out[i,:] = dinv[i]^2 * h[i,:]   (self loop term);  Dq = D/4, dqs = log2(Dq)
__global__ void k_prop_init(const float* __restrict__ h, int ldh,
                            const float* __restrict__ dinv,
                            float* __restrict__ out, int ldo, int dqs, int n) {
    int t = blockIdx.x * 256 + threadIdx.x;
    int i = t >> dqs, c = (t & ((1 << dqs) - 1)) << 2;
    if (i >= n) return;
    float w = dinv[i];
    w *= w;
    float4 hv = *(const float4*)&h[(size_t)i * ldh + c];
    float4 o = {w * hv.x, w * hv.y, w * hv.z, w * hv.w};
    *(float4*)&out[(size_t)i * ldo + c] = o;
}
// out[dst,:] += dinv[src]*dinv[dst] * h[src,:]
__global__ void k_prop_edges(const int* __restrict__ src, const int* __restrict__ dst,
                             const float* __restrict__ dinv,
                             const float* __restrict__ h, int ldh,
                             float* __restrict__ out, int ldo, int dqs, int e) {
    int t = blockIdx.x * 256 + threadIdx.x;
    int ed = t >> dqs, c = (t & ((1 << dqs) - 1)) << 2;
    if (ed >= e) return;
    int s = src[ed], d0 = dst[ed];
    float w = dinv[s] * dinv[d0];
    const float4 hv = *(const float4*)&h[(size_t)s * ldh + c];
    float* o = &out[(size_t)d0 * ldo + c];
    unsafeAtomicAdd(o + 0, w * hv.x);
    unsafeAtomicAdd(o + 1, w * hv.y);
    unsafeAtomicAdd(o + 2, w * hv.z);
    unsafeAtomicAdd(o + 3, w * hv.w);
}

// ---------------------------------------------------------------- WMMA GEMM
// C[m, ccol0+n] = act( sum_k A[m,k] * Wt[n,k] + bias[n] )
// A: fp32, rows beyond M zero-padded by caller.  Wt: bf16 n-major [Nout][K], K%32==0.
// block = 256 threads (8 waves), tile 128(M) x 64(N), K step 32, double-buffered LDS.
// A staged via b128 global loads + v_perm bf16 pack + b64 LDS stores;
// B staged via CDNA5 global_load_async_to_lds_b128 (ASYNCcnt).
#define BM 128
#define BN 64
#define PITCH 40   // ushort pitch: 80B = 20 banks -> conflict-free + 16B-aligned fragments

template <int ACT>
__global__ __launch_bounds__(256)
void k_gemm(const float* __restrict__ A, int lda, int M, int K,
            const u16* __restrict__ Wt, const float* __restrict__ bias,
            float* __restrict__ C, int ldc, int ccol0) {
    __shared__ u16 sA[2][BM * PITCH];
    __shared__ u16 sB[2][BN * PITCH];

    const int tid  = threadIdx.x;
    const int wv   = tid >> 5;
    const int lane = tid & 31;
    const int half = lane >> 4;
    const int l16  = lane & 15;
    const int bm   = blockIdx.x * BM;
    const int bn   = blockIdx.y * BN;

    // staging coordinates (guard-free, fully vectorized)
    const int ar = tid >> 3;          // A row 0..31 (+32*i)
    const int ac = (tid & 7) << 2;    // A k offset 0,4,...,28
    const int br = tid >> 2;          // B n row 0..63
    const int bc = (tid & 3) << 3;    // B k offset 0,8,16,24

    const float* Abase = A + (size_t)bm * lda;
    const u16*   Bbase = Wt + (size_t)bn * K;

    float4 a4[4];

    auto loadgA = [&](int kb) {
#pragma unroll
        for (int i = 0; i < 4; ++i)
            a4[i] = *(const float4*)(Abase + (size_t)(ar + 32 * i) * lda + kb + ac);
    };
    auto stldsA = [&](int buf) {
#pragma unroll
        for (int i = 0; i < 4; ++i) {
            uint2 pk;
            pk.x = pack_bf16(a4[i].x, a4[i].y);
            pk.y = pack_bf16(a4[i].z, a4[i].w);
            *(uint2*)&sA[buf][(ar + 32 * i) * PITCH + ac] = pk;
        }
    };
    // async copy of one b128 per lane: 256 threads cover the 32x64 bf16 B tile
    auto loadB_async = [&](int buf, int kb) {
        u32 ldsoff = (u32)(uintptr_t)&sB[buf][br * PITCH + bc];
        const u16* g = Bbase + (size_t)br * K + kb + bc;
        asm volatile("global_load_async_to_lds_b128 %0, %1, off"
                     :: "v"(ldsoff), "v"(g)
                     : "memory");
    };

    v8f acc[4];
#pragma unroll
    for (int f = 0; f < 4; ++f) {
        v8f z = {0.f, 0.f, 0.f, 0.f, 0.f, 0.f, 0.f, 0.f};
        acc[f] = z;
    }

    const int nkb = K >> 5;
    loadB_async(0, 0);
    loadgA(0);
    stldsA(0);
    asm volatile("s_wait_asynccnt 0" ::: "memory");
    __syncthreads();

    for (int ib = 0; ib < nkb; ++ib) {
        const int  cur = ib & 1;
        const bool nx  = (ib + 1 < nkb);
        if (nx) {
            loadB_async((ib + 1) & 1, (ib + 1) << 5);
            loadgA((ib + 1) << 5);
        }

        // A fragment (16x32 bf16): 2x b128 from LDS (16B-aligned thanks to PITCH=40)
        union { uint4 q[2]; u32 u[8]; v16bf v; } af;
        const uint4* pa = (const uint4*)&sA[cur][(wv * 16 + l16) * PITCH + 8 * half];
        af.q[0] = pa[0];
        af.q[1] = pa[2];   // +32 bytes: K dwords {16,18,20,22}

#pragma unroll
        for (int f = 0; f < 4; ++f) {
            union { uint4 q[2]; u32 u[8]; v16bf v; } bfm;
            const uint4* pb = (const uint4*)&sB[cur][(f * 16 + l16) * PITCH + 8 * half];
            bfm.q[0] = pb[0];
            bfm.q[1] = pb[2];
            acc[f] = __builtin_amdgcn_wmma_f32_16x16x32_bf16(
                false, af.v, false, bfm.v, (short)0, acc[f], false, false);
        }

        if (nx) stldsA((ib + 1) & 1);
        asm volatile("s_wait_asynccnt 0" ::: "memory");
        __syncthreads();
    }

    // epilogue: C/D layout — vgpr r, lane: N=l16, M=r+8*half
#pragma unroll
    for (int f = 0; f < 4; ++f) {
        int n = bn + f * 16 + l16;
        float bv = bias[n];
#pragma unroll
        for (int r = 0; r < 8; ++r) {
            int row = bm + wv * 16 + half * 8 + r;
            if (row < M) {
                float v = acc[f][r] + bv;
                if (ACT) v = gelu_f(v);
                C[(size_t)row * ldc + ccol0 + n] = v;
            }
        }
    }
}

// ---------------------------------------------------------------- LayerNorm(1024) + gelu
__global__ __launch_bounds__(256)
void k_ln_gelu(const float* __restrict__ X, int ldx,
               const float* __restrict__ g, const float* __restrict__ b,
               float* __restrict__ Y, int ldy) {
    const int row = blockIdx.x;
    const int tid = threadIdx.x;
    __shared__ float rs[8], rs2[8];
    float4 x = *(const float4*)&X[(size_t)row * ldx + tid * 4];
    float s  = x.x + x.y + x.z + x.w;
    float s2 = x.x * x.x + x.y * x.y + x.z * x.z + x.w * x.w;
    for (int off = 16; off; off >>= 1) {
        s  += __shfl_xor(s, off, 32);
        s2 += __shfl_xor(s2, off, 32);
    }
    if ((tid & 31) == 0) { rs[tid >> 5] = s; rs2[tid >> 5] = s2; }
    __syncthreads();
    float ts = 0.f, ts2 = 0.f;
#pragma unroll
    for (int i = 0; i < 8; ++i) { ts += rs[i]; ts2 += rs2[i]; }
    float mu  = ts * (1.0f / 1024.0f);
    float var = ts2 * (1.0f / 1024.0f) - mu * mu;
    float inv = rsqrtf(var + 1e-5f);
    float4 gv = *(const float4*)&g[tid * 4];
    float4 bv = *(const float4*)&b[tid * 4];
    float4 y;
    y.x = gelu_f((x.x - mu) * inv * gv.x + bv.x);
    y.y = gelu_f((x.y - mu) * inv * gv.y + bv.y);
    y.z = gelu_f((x.z - mu) * inv * gv.z + bv.z);
    y.w = gelu_f((x.w - mu) * inv * gv.w + bv.w);
    *(float4*)&Y[(size_t)row * ldy + tid * 4] = y;
}

// ---------------------------------------------------------------- final head: [N,128] @ [128,1] + b
__global__ __launch_bounds__(256)
void k_head(const float* __restrict__ Hb, int ldh,
            const float* __restrict__ w3, const float* __restrict__ b3,
            float* __restrict__ out, int M) {
    int wv = threadIdx.x >> 5, lane = threadIdx.x & 31;
    int row = blockIdx.x * 8 + wv;
    if (row >= M) return;
    float s = 0.f;
#pragma unroll
    for (int i = 0; i < 4; ++i) {
        int c = lane + i * 32;
        s += Hb[(size_t)row * ldh + c] * w3[c];
    }
    for (int off = 16; off; off >>= 1) s += __shfl_xor(s, off, 32);
    if (lane == 0) out[row] = s + b3[0];
}

// ================================================================ host
static inline int cdiv(int a, int b) { return (a + b - 1) / b; }

extern "C" void kernel_launch(void* const* d_in, const int* in_sizes, int n_in,
                              void* d_out, int out_size, void* d_ws, size_t ws_size,
                              hipStream_t stream) {
    const float* x      = (const float*)d_in[0];
    const int*   ei     = (const int*)d_in[1];
    const float* w_in   = (const float*)d_in[2];
    const float* b_in   = (const float*)d_in[3];
    const float* mh_w0  = (const float*)d_in[4];
    const float* mh_b0  = (const float*)d_in[5];
    const float* mh_w12 = (const float*)d_in[6];
    const float* mh_b12 = (const float*)d_in[7];
    const float* ln_g   = (const float*)d_in[8];
    const float* ln_b   = (const float*)d_in[9];
    const float* w1     = (const float*)d_in[10];
    const float* b1     = (const float*)d_in[11];
    const float* w2     = (const float*)d_in[12];
    const float* b2     = (const float*)d_in[13];
    const float* w3     = (const float*)d_in[14];
    const float* b3     = (const float*)d_in[15];
    float* out = (float*)d_out;

    const int* srcp = ei;
    const int* dstp = ei + EE;

    // workspace carve-out
    size_t off = 0;
    auto take = [&](size_t bytes) {
        size_t o = off;
        off += (bytes + 255) & ~(size_t)255;
        return o;
    };
    char* ws = (char*)d_ws;
    float* dinv    = (float*)(ws + take((size_t)NN * 4));
    float* Xp      = (float*)(ws + take((size_t)MPAD * KIN * 4));
    u16* w_in_t    = (u16*)(ws + take((size_t)HH * KIN * 2));
    u16* mh_w0_t   = (u16*)(ws + take((size_t)PP * HH * HH * 2));
    u16* mh_w12_t  = (u16*)(ws + take((size_t)2 * PP * HH * PH * 2));
    u16* w1_t      = (u16*)(ws + take((size_t)HH * PH * 2));
    u16* w2_t      = (u16*)(ws + take((size_t)(HH / 2) * HH * 2));
    float* Hc      = (float*)(ws + take((size_t)MPAD * PH * 4));
    float* Pa      = (float*)(ws + take((size_t)MPAD * PH * 4));
    float* Pb      = (float*)(ws + take((size_t)MPAD * PH * 4));
    float* Cc      = (float*)(ws + take((size_t)MPAD * PH * 4));
    (void)ws_size;

    const int PAD256  = (MPAD - NN) * HH;   // ld-256 pad floats
    const int PAD1024 = (MPAD - NN) * PH;   // ld-1024 pad floats

    // 0) zero the pad rows that GEMM A-tiles will read (guard-free loads)
    k_zero<<<cdiv(PAD256, 256), 256, 0, stream>>>(Hc + (size_t)NN * HH, PAD256);
    k_zero<<<cdiv(PAD256, 256), 256, 0, stream>>>(Pa + (size_t)NN * HH, PAD256);
    k_zero<<<cdiv(PAD256, 256), 256, 0, stream>>>(Pb + (size_t)NN * HH, PAD256);
    k_zero<<<cdiv(PAD1024, 256), 256, 0, stream>>>(Hc + (size_t)NN * PH, PAD1024);
    k_zero<<<cdiv(PAD1024, 256), 256, 0, stream>>>(Pa + (size_t)NN * PH, PAD1024);
    k_zero<<<cdiv(PAD1024, 256), 256, 0, stream>>>(Pb + (size_t)NN * PH, PAD1024);
    k_zero<<<cdiv(PAD1024, 256), 256, 0, stream>>>(Cc + (size_t)NN * PH, PAD1024);

    // 1) gcn_norm: deg (with self loop) -> dinv
    k_fill1<<<cdiv(NN, 256), 256, 0, stream>>>(dinv, NN);
    k_degcnt<<<cdiv(EE, 256), 256, 0, stream>>>(dstp, dinv, EE);
    k_rsqrt<<<cdiv(NN, 256), 256, 0, stream>>>(dinv, NN);

    // 2) pack x, weights -> transposed bf16 (n-major [Nout][K])
    k_pack_x<<<cdiv(MPAD * KIN, 256), 256, 0, stream>>>(x, Xp);
    k_cvt_tr<<<dim3(cdiv(KIN, 256), HH), 256, 0, stream>>>(w_in, w_in_t, F_IN, KIN, HH);
    for (int p = 0; p < PP; ++p)
        k_cvt_tr<<<dim3(cdiv(HH, 256), HH), 256, 0, stream>>>(
            mh_w0 + (size_t)p * HH * HH, mh_w0_t + (size_t)p * HH * HH, HH, HH, HH);
    for (int lp = 0; lp < 2 * PP; ++lp)
        k_cvt_tr<<<dim3(cdiv(PH, 256), HH), 256, 0, stream>>>(
            mh_w12 + (size_t)lp * PH * HH, mh_w12_t + (size_t)lp * HH * PH, PH, PH, HH);
    k_cvt_tr<<<dim3(cdiv(PH, 256), HH), 256, 0, stream>>>(w1, w1_t, PH, PH, HH);
    k_cvt_tr<<<dim3(cdiv(HH, 256), HH / 2), 256, 0, stream>>>(w2, w2_t, HH, HH, HH / 2);

    const int gx = MPAD / BM;   // 79

    // 3) h = gelu(x @ w_in + b_in)  -> Hc [N,256] (ld 256)
    k_gemm<1><<<dim3(gx, HH / BN), 256, 0, stream>>>(Xp, KIN, NN, KIN,
                                                     w_in_t, b_in, Hc, HH, 0);

    // 4) mixhop layers
    const float* hin = Hc;
    int ldh = HH;
    float* pbuf[2] = {Pa, Pb};
    for (int l = 0; l < 3; ++l) {
        const int D = (l == 0) ? HH : PH;
        const int dqs = (l == 0) ? 6 : 8;   // log2(D/4)
        const u16* Wl = (l == 0) ? mh_w0_t : (mh_w12_t + (size_t)(l - 1) * PP * HH * PH);
        const float* Bl = (l == 0) ? mh_b0 : (mh_b12 + (size_t)(l - 1) * PP * HH);
        const size_t wstep = (size_t)D * HH;

        // p = 0
        k_gemm<0><<<dim3(gx, HH / BN), 256, 0, stream>>>(hin, ldh, NN, D,
                                                         Wl, Bl, Cc, PH, 0);
        const float* cur = hin;
        int ldcur = ldh;
        for (int p = 1; p < PP; ++p) {
            float* nxt = pbuf[(p - 1) & 1];
            k_prop_init<<<cdiv(NN * (D / 4), 256), 256, 0, stream>>>(
                cur, ldcur, dinv, nxt, D, dqs, NN);
            k_prop_edges<<<cdiv(EE * (D / 4), 256), 256, 0, stream>>>(
                srcp, dstp, dinv, cur, ldcur, nxt, D, dqs, EE);
            k_gemm<0><<<dim3(gx, HH / BN), 256, 0, stream>>>(
                nxt, D, NN, D, Wl + (size_t)p * wstep, Bl + p * HH, Cc, PH, p * HH);
            cur = nxt;
            ldcur = D;
        }
        // layernorm(1024) + gelu -> Hc (ld 1024)
        k_ln_gelu<<<NN, 256, 0, stream>>>(Cc, PH, ln_g + (size_t)l * PH,
                                          ln_b + (size_t)l * PH, Hc, PH);
        hin = Hc;
        ldh = PH;
    }

    // 5) MLP head
    k_gemm<1><<<dim3(gx, HH / BN), 256, 0, stream>>>(Hc, PH, NN, PH, w1_t, b1, Cc, PH, 0);
    k_gemm<1><<<dim3(gx, (HH / 2) / BN), 256, 0, stream>>>(Cc, PH, NN, HH, w2_t, b2, Pa, PH, 0);
    k_head<<<cdiv(NN, 8), 256, 0, stream>>>(Pa, PH, w3, b3, out, NN);
}
